// CapsuleNetwork_19387482374166
// MI455X (gfx1250) — matrix operand order repaired and checked by reading, not compiled, not run on hardware
//
#include <hip/hip_runtime.h>

// CDNA5 / gfx1250: wave32, WMMA 16x16x32 bf16, f32 accumulate.
typedef __attribute__((ext_vector_type(16))) __bf16 v16bf;
typedef __attribute__((ext_vector_type(8)))  float  v8f;

#define S_DIM 256
#define I_DIM 8
#define D_DIM 128

// LDS strides (elements). +8 bf16 padding => row stride % 64 DWORDs == 4:
// the 16 lanes of a fragment load hit distinct banks; rows stay 16B aligned.
#define HT_STRIDE   264   // H transposed [128][264] bf16
#define HR_STRIDE   136   // H row-major  [256][136] bf16
#define W_STRIDE    264   // w            [16][264]  bf16 (rows 8..15 zero)
#define CAPS_STRIDE 136   // caps bf16    [16][136]  (rows 8..15 zero)

union Frag {
    uint4 q[2];
    v16bf v;
};

__device__ __forceinline__ unsigned short f2bf(float x) {
    unsigned int u = __float_as_uint(x);
    unsigned int r = (u + 0x7FFFu + ((u >> 16) & 1u)) >> 16;  // RNE
    return (unsigned short)r;
}

__device__ __forceinline__ unsigned int pack_bf2(float lo, float hi) {
    return (unsigned int)f2bf(lo) | ((unsigned int)f2bf(hi) << 16);
}

__global__ __launch_bounds__(256, 1)
void capsule_routing_kernel(const float* __restrict__ hist,
                            const float* __restrict__ cw_g,
                            const int*   __restrict__ mask_g,
                            float*       __restrict__ out) {
    // 162,336 B static LDS = 80 x 2KiB granules = 160 KiB:
    // exactly two workgroups co-reside per 320 KiB WGP (stage/compute overlap).
    __shared__ __align__(16) unsigned short s_ht[D_DIM * HT_STRIDE];   // 67584 B
    __shared__ __align__(16) unsigned short s_hr[S_DIM * HR_STRIDE];   // 69632 B
    __shared__ __align__(16) unsigned short s_w[16 * W_STRIDE];        //  8448 B
    __shared__ __align__(16) unsigned short s_caps[16 * CAPS_STRIDE];  //  4352 B
    __shared__ __align__(16) float s_cw[I_DIM * S_DIM];                //  8192 B
    __shared__ __align__(16) float s_cap[I_DIM * D_DIM];               //  4096 B
    __shared__ float s_scalar[I_DIM];

    const int b    = blockIdx.x;
    const int t    = threadIdx.x;
    const int lane = t & 31;
    const int wave = t >> 5;        // 8 waves
    const int hi   = lane >> 4;     // half-wave select (K range / M+8)
    const int lh   = lane & 15;

    // ---- Stage H (256x128 f32 -> bf16, both orientations), cw, zero-pad w ----
    // Each thread handles rows (s0, s0+1) x 4 columns per iteration so the
    // transposed copy is written as packed b32 stores (half the DS stores).
    const float4* hsrc = (const float4*)(hist + (size_t)b * S_DIM * D_DIM);
    const int q  = lane;            // float4 column 0..31 (d0 = 4q)
    const int d0 = q << 2;
    #pragma unroll
    for (int j = 0; j < 16; ++j) {
        const int s0 = (j * 8 + wave) * 2;     // even row
        float4 va = hsrc[s0 * 32 + q];          // row s0, 512B contiguous/wave
        float4 vb = hsrc[(s0 + 1) * 32 + q];    // row s0+1
        // row-major copy (two b64 stores, conflict-free: lanes span one row)
        unsigned int* pa = (unsigned int*)&s_hr[s0 * HR_STRIDE + d0];
        pa[0] = pack_bf2(va.x, va.y);
        pa[1] = pack_bf2(va.z, va.w);
        unsigned int* pb = (unsigned int*)&s_hr[(s0 + 1) * HR_STRIDE + d0];
        pb[0] = pack_bf2(vb.x, vb.y);
        pb[1] = pack_bf2(vb.z, vb.w);
        // transposed copy: pack (s0, s0+1) pairs -> 4 b32 stores
        *(unsigned int*)&s_ht[(d0 + 0) * HT_STRIDE + s0] = pack_bf2(va.x, vb.x);
        *(unsigned int*)&s_ht[(d0 + 1) * HT_STRIDE + s0] = pack_bf2(va.y, vb.y);
        *(unsigned int*)&s_ht[(d0 + 2) * HT_STRIDE + s0] = pack_bf2(va.z, vb.z);
        *(unsigned int*)&s_ht[(d0 + 3) * HT_STRIDE + s0] = pack_bf2(va.w, vb.w);
    }
    const float* csrc = cw_g + (size_t)b * I_DIM * S_DIM;
    #pragma unroll
    for (int j = 0; j < 8; ++j) {
        int idx = t + j * 256;
        s_cw[idx] = csrc[idx];
    }
    for (int idx = t; idx < 8 * W_STRIDE; idx += 256)
        s_w[8 * W_STRIDE + idx] = 0;        // A-matrix pad rows (M=8..15)
    const int my_mask = mask_g[(size_t)b * S_DIM + t];
    __syncthreads();

    for (int iter = 0; iter < 3; ++iter) {
        // ---- softmax over I per column s = t, then mask, store bf16 ----
        {
            float v[8], mx = -3.0e38f;
            #pragma unroll
            for (int i = 0; i < 8; ++i) { v[i] = s_cw[i * S_DIM + t]; mx = fmaxf(mx, v[i]); }
            float sum = 0.f;
            #pragma unroll
            for (int i = 0; i < 8; ++i) { v[i] = __expf(v[i] - mx); sum += v[i]; }
            float inv = __frcp_rn(sum);
            #pragma unroll
            for (int i = 0; i < 8; ++i) {
                float wv = (my_mask == 0) ? 0.f : v[i] * inv;
                s_w[i * W_STRIDE + t] = f2bf(wv);
            }
        }
        __syncthreads();

        // ---- forward GEMM: cap(16x128) = w(16x256) @ H(256x128); wave = N-tile ----
        Frag a, bb;
        v8f acc = {0.f, 0.f, 0.f, 0.f, 0.f, 0.f, 0.f, 0.f};
        #pragma unroll
        for (int ks = 0; ks < 8; ++ks) {
            const unsigned short* ap = &s_w[lh * W_STRIDE + ks * 32 + hi * 8];
            a.q[0] = *(const uint4*)ap;
            a.q[1] = *(const uint4*)(ap + 16);
            const unsigned short* bp = &s_ht[(wave * 16 + lh) * HT_STRIDE + ks * 32 + hi * 8];
            bb.q[0] = *(const uint4*)bp;
            bb.q[1] = *(const uint4*)(bp + 16);
            acc = __builtin_amdgcn_wmma_f32_16x16x32_bf16(
                false, a.v, false, bb.v, (short)0, acc, false, false);
        }
        if (hi == 0) {                      // rows M=0..7 are the real capsules
            #pragma unroll
            for (int r = 0; r < 8; ++r)
                s_cap[r * D_DIM + wave * 16 + lh] = acc[r];
        }
        __syncthreads();

        // ---- squash scalar: wave i reduces ||cap[i]||^2 over 128 cols ----
        {
            const float4 c4 = *(const float4*)&s_cap[wave * D_DIM + lane * 4];
            float p = c4.x * c4.x + c4.y * c4.y + c4.z * c4.z + c4.w * c4.w;
            #pragma unroll
            for (int off = 16; off > 0; off >>= 1) p += __shfl_xor(p, off, 32);
            if (lane == 0) {
                s_scalar[wave] = p / (1.0f + p) * __frsqrt_rn(p + 1e-9f);
            }
        }
        __syncthreads();

        if (iter == 2) break;

        // ---- caps_bf16 = scalar[i] * cap (rows 8..15 zero) ----
        for (int idx = t; idx < 16 * D_DIM; idx += 256) {
            int i = idx >> 7, d = idx & 127;
            float val = (i < 8) ? s_scalar[i] * s_cap[i * D_DIM + d] : 0.f;
            s_caps[i * CAPS_STRIDE + d] = f2bf(val);
        }
        __syncthreads();

        // ---- delta GEMM: cw(8x256) += caps(16x128) @ H^T(128x256); 2 N-tiles/wave ----
        #pragma unroll
        for (int sub = 0; sub < 2; ++sub) {
            const int nt = wave * 2 + sub;
            v8f acc2 = {0.f, 0.f, 0.f, 0.f, 0.f, 0.f, 0.f, 0.f};
            #pragma unroll
            for (int ks = 0; ks < 4; ++ks) {
                const unsigned short* ap = &s_caps[lh * CAPS_STRIDE + ks * 32 + hi * 8];
                a.q[0] = *(const uint4*)ap;
                a.q[1] = *(const uint4*)(ap + 16);
                const unsigned short* bp = &s_hr[(nt * 16 + lh) * HR_STRIDE + ks * 32 + hi * 8];
                bb.q[0] = *(const uint4*)bp;
                bb.q[1] = *(const uint4*)(bp + 16);
                acc2 = __builtin_amdgcn_wmma_f32_16x16x32_bf16(
                    false, a.v, false, bb.v, (short)0, acc2, false, false);
            }
            if (hi == 0) {                  // unique (i, s) per lane -> plain RMW
                const int scol = nt * 16 + lh;
                #pragma unroll
                for (int r = 0; r < 8; ++r)
                    s_cw[r * S_DIM + scol] += acc2[r];
            }
        }
        __syncthreads();
    }

    // ---- output: out[b,i,d] = scalar[i] * cap[i,d] (coalesced f32 stores) ----
    #pragma unroll
    for (int j = 0; j < 4; ++j) {
        int idx = t + j * 256;              // 0..1023
        int i   = idx >> 7;
        out[(size_t)b * (I_DIM * D_DIM) + idx] = s_scalar[i] * s_cap[idx];
    }
}

extern "C" void kernel_launch(void* const* d_in, const int* in_sizes, int n_in,
                              void* d_out, int out_size, void* d_ws, size_t ws_size,
                              hipStream_t stream) {
    const float* hist  = (const float*)d_in[0];   // (1024, 256, 128) f32
    const float* cw    = (const float*)d_in[1];   // (1024, 8, 256)  f32
    const int*   mask  = (const int*)d_in[2];     // (1024, 256)     i32
    float*       out   = (float*)d_out;           // (1024, 8, 128)  f32
    (void)in_sizes; (void)n_in; (void)out_size; (void)d_ws; (void)ws_size;

    capsule_routing_kernel<<<dim3(1024), dim3(256), 0, stream>>>(hist, cw, mask, out);
}
